// ScaledDotProductAttention_84250078478576
// MI455X (gfx1250) — matrix operand order repaired and checked by reading, not compiled, not run on hardware
//
#include <hip/hip_runtime.h>
#include <hip/hip_bf16.h>

// ---------------------------------------------------------------------------
// Scaled dot-product attention for gfx1250 (MI455X), fp32 in/out.
//   scores  = Q @ K^T            (8192 x 8192)
//   attn    = softmax(scores,1)  (written to d_out, second output)
//   context = attn @ V           (8192 x 1024, first output)
// Matrix math runs on V_WMMA_F32_16X16X32_BF16 (f32 accumulate), the CDNA5
// high-throughput path; softmax is a register-resident row pass.
// ---------------------------------------------------------------------------

#define LQ 8192
#define LK 8192
#define DD 1024
#define DVV 1024

typedef __attribute__((ext_vector_type(16))) __bf16 v16bf;
typedef __attribute__((ext_vector_type(8)))  float  v8f;

// ---- WMMA operand loaders (contiguous k order k = 16*(lane/16) + s, A and B alike, wave32) ---------

// A-matrix 16x32 bf16: lane l (0..15) row M=l holds K 0..15 (elems 0..15);
// lanes 16..31 hold K 16..31.
__device__ __forceinline__ v16bf load_a_bf16(const float* __restrict__ base,
                                             int ld, int mbase, int k0, int lane) {
    const int half = lane >> 4;            // 0: lanes 0-15, 1: lanes 16-31
    const int m    = mbase + (lane & 15);
    const float* p0 = base + (size_t)m * ld + k0 + half * 16;
    const float* p1 = p0 + 8;
    v16bf a;
#pragma unroll
    for (int j = 0; j < 8; ++j) {
        a[j]     = (__bf16)p0[j];
        a[j + 8] = (__bf16)p1[j];
    }
    return a;
}

// B-matrix 32x16 bf16 where the logical B is row-major SRC transposed:
// B[k][n] = M[n][k]  (used for Q @ K^T).  Lane l holds column n=l%16,
// K 0..15 (lanes 0-15) or K 16..31 (lanes 16-31): a contiguous 64B row run.
__device__ __forceinline__ v16bf load_bT_bf16(const float* __restrict__ base,
                                              int ld, int nbase, int k0, int lane) {
    const int half = lane >> 4;
    const int n    = nbase + (lane & 15);
    const float* p = base + (size_t)n * ld + k0 + half * 16;
    v16bf b;
#pragma unroll
    for (int j = 0; j < 16; ++j) b[j] = (__bf16)p[j];
    return b;
}

// B-matrix 32x16 bf16 from a genuinely row-major [K x N] operand (used for
// attn @ V): B[k][n] = M[k0+k][nbase+n]; lane l column n=l%16, stride-ld reads.
__device__ __forceinline__ v16bf load_bRM_bf16(const float* __restrict__ base,
                                               int ld, int k0, int nbase, int lane) {
    const int half = lane >> 4;
    const int n    = nbase + (lane & 15);
    const float* p = base + (size_t)(k0 + half * 16) * ld + n;
    v16bf b;
#pragma unroll
    for (int j = 0; j < 16; ++j) b[j] = (__bf16)p[(size_t)j * ld];
    return b;
}

// C/D 16x16 f32: elem r in lane l -> [mbase + r + 8*(l>=16)][nbase + l%16]
__device__ __forceinline__ void store_c(float* __restrict__ base, int ld,
                                        int mbase, int nbase, v8f c, int lane) {
    const int half = lane >> 4;
    const int n    = nbase + (lane & 15);
#pragma unroll
    for (int r = 0; r < 8; ++r)
        base[(size_t)(mbase + r + half * 8) * ld + n] = c[r];
}

// ---------------------------------------------------------------------------
// Kernel 1: S = Q @ K^T.  Wave tile 64x64 (16 accumulators), block = 8 waves
// (2 x 4) -> 128x256 tile.  Grid (LK/256, LQ/128) = (32, 64).
// ---------------------------------------------------------------------------
__global__ __launch_bounds__(256) void qk_scores_kernel(
    const float* __restrict__ Q, const float* __restrict__ Km,
    float* __restrict__ S) {
    const int lane  = threadIdx.x & 31;
    const int wave  = threadIdx.x >> 5;
    const int wm    = wave >> 2;           // 0..1
    const int wn    = wave & 3;            // 0..3
    const int mbase = blockIdx.y * 128 + wm * 64;
    const int nbase = blockIdx.x * 256 + wn * 64;

    v8f acc[4][4] = {};

    for (int k0 = 0; k0 < DD; k0 += 32) {
        v16bf a[4];
#pragma unroll
        for (int i = 0; i < 4; ++i)
            a[i] = load_a_bf16(Q, DD, mbase + 16 * i, k0, lane);
#pragma unroll
        for (int t = 0; t < 4; ++t) {
            v16bf b = load_bT_bf16(Km, DD, nbase + 16 * t, k0, lane);
#pragma unroll
            for (int i = 0; i < 4; ++i)
                acc[i][t] = __builtin_amdgcn_wmma_f32_16x16x32_bf16(
                    false, a[i], false, b, (short)0, acc[i][t], false, false);
        }
    }

#pragma unroll
    for (int i = 0; i < 4; ++i)
#pragma unroll
        for (int t = 0; t < 4; ++t)
            store_c(S, LK, mbase + 16 * i, nbase + 16 * t, acc[i][t], lane);
}

// ---------------------------------------------------------------------------
// Kernel 2: in-place row softmax over LK=8192.  One block per row, 256
// threads x 32 elements held in registers -> exactly one read + one write of
// the 268MB matrix.  wave32 shuffle reduce + LDS cross-wave reduce.
// ---------------------------------------------------------------------------
__global__ __launch_bounds__(256) void softmax_rows_kernel(float* __restrict__ S) {
    float* row = S + (size_t)blockIdx.x * LK;
    const int tid  = threadIdx.x;
    const int lane = tid & 31;
    const int wave = tid >> 5;

    float v[32];
    float m = -3.402823466e38f;
#pragma unroll
    for (int i = 0; i < 32; ++i) {
        v[i] = row[i * 256 + tid];
        m = fmaxf(m, v[i]);
    }
#pragma unroll
    for (int off = 16; off > 0; off >>= 1)
        m = fmaxf(m, __shfl_xor(m, off, 32));

    __shared__ float smax[8];
    __shared__ float ssum[8];
    if (lane == 0) smax[wave] = m;
    __syncthreads();
#pragma unroll
    for (int w = 0; w < 8; ++w) m = fmaxf(m, smax[w]);

    float sum = 0.0f;
#pragma unroll
    for (int i = 0; i < 32; ++i) {
        v[i] = __expf(v[i] - m);
        sum += v[i];
    }
#pragma unroll
    for (int off = 16; off > 0; off >>= 1)
        sum += __shfl_xor(sum, off, 32);
    if (lane == 0) ssum[wave] = sum;
    __syncthreads();
    sum = 0.0f;
#pragma unroll
    for (int w = 0; w < 8; ++w) sum += ssum[w];

    const float inv = 1.0f / sum;
#pragma unroll
    for (int i = 0; i < 32; ++i)
        row[i * 256 + tid] = v[i] * inv;
}

// ---------------------------------------------------------------------------
// Kernel 3: context = P @ V.  Wave tile 32x64 (8 accumulators), block = 8
// waves (2 x 4) -> 64x256 tile.  Grid (DVV/256, LQ/64) = (4, 128).  K-loop
// streams P once from HBM; V (32MB) stays L2-resident.
// ---------------------------------------------------------------------------
__global__ __launch_bounds__(256) void pv_context_kernel(
    const float* __restrict__ P, const float* __restrict__ V,
    float* __restrict__ Ctx) {
    const int lane  = threadIdx.x & 31;
    const int wave  = threadIdx.x >> 5;
    const int wm    = wave >> 2;           // 0..1
    const int wn    = wave & 3;            // 0..3
    const int mbase = blockIdx.y * 64  + wm * 32;
    const int nbase = blockIdx.x * 256 + wn * 64;

    v8f acc[2][4] = {};

    for (int k0 = 0; k0 < LK; k0 += 32) {
        v16bf a[2];
#pragma unroll
        for (int i = 0; i < 2; ++i)
            a[i] = load_a_bf16(P, LK, mbase + 16 * i, k0, lane);
#pragma unroll
        for (int t = 0; t < 4; ++t) {
            v16bf b = load_bRM_bf16(V, DVV, k0, nbase + 16 * t, lane);
#pragma unroll
            for (int i = 0; i < 2; ++i)
                acc[i][t] = __builtin_amdgcn_wmma_f32_16x16x32_bf16(
                    false, a[i], false, b, (short)0, acc[i][t], false, false);
        }
    }

#pragma unroll
    for (int i = 0; i < 2; ++i)
#pragma unroll
        for (int t = 0; t < 4; ++t)
            store_c(Ctx, DVV, mbase + 16 * i, nbase + 16 * t, acc[i][t], lane);
}

// ---------------------------------------------------------------------------
extern "C" void kernel_launch(void* const* d_in, const int* in_sizes, int n_in,
                              void* d_out, int out_size, void* d_ws, size_t ws_size,
                              hipStream_t stream) {
    (void)in_sizes; (void)n_in; (void)out_size; (void)d_ws; (void)ws_size;
    const float* q = (const float*)d_in[0];
    const float* k = (const float*)d_in[1];
    const float* v = (const float*)d_in[2];

    float* ctx  = (float*)d_out;                       // [LQ, DVV]
    float* attn = (float*)d_out + (size_t)LQ * DVV;    // [LQ, LK]

    // 1) raw scores into the attention output buffer
    qk_scores_kernel<<<dim3(LK / 256, LQ / 128), 256, 0, stream>>>(q, k, attn);
    // 2) in-place row softmax
    softmax_rows_kernel<<<dim3(LQ), 256, 0, stream>>>(attn);
    // 3) context = attn @ V
    pv_context_kernel<<<dim3(DVV / 256, LQ / 64), 256, 0, stream>>>(attn, v, ctx);
}
